// AttentionalGraphAggregation_83270825935257
// MI455X (gfx1250) — matrix operand order repaired and checked by reading, not compiled, not run on hardware
//
#include <hip/hip_runtime.h>
#include <math.h>

// ---------------------------------------------------------------------------
// Types for CDNA5 WMMA (wave32): v_wmma_f32_16x16x32_f16
// ---------------------------------------------------------------------------
typedef _Float16 v16h   __attribute__((ext_vector_type(16)));
typedef _Float16 halfx8 __attribute__((ext_vector_type(8)));
typedef float    v8f    __attribute__((ext_vector_type(8)));

#define N_ROWS   1000000
#define IN_CH    256
#define OUT_CH   256
#define HID      128
#define SEGS     16384
#define M_TILES  (N_ROWS / 16)   // 62500

// Monotonic float <-> ordered-uint mapping for atomicMax on floats.
__device__ __forceinline__ unsigned flipf(float f) {
    unsigned u = __float_as_uint(f);
    return (u & 0x80000000u) ? ~u : (u | 0x80000000u);
}
__device__ __forceinline__ float unflipf(unsigned u) {
    return __uint_as_float((u & 0x80000000u) ? (u & 0x7fffffffu) : ~u);
}

// ---------------------------------------------------------------------------
// Kernel 0: zero the output / denom, sentinel the segment-max array.
// ---------------------------------------------------------------------------
__global__ void init_kernel(float* __restrict__ out,
                            float* __restrict__ denom,
                            unsigned* __restrict__ smaxu) {
    int i = blockIdx.x * blockDim.x + threadIdx.x;
    int stride = gridDim.x * blockDim.x;
    for (int j = i; j < SEGS * OUT_CH; j += stride) out[j] = 0.0f;
    for (int j = i; j < SEGS; j += stride) { denom[j] = 0.0f; smaxu[j] = 0u; }
}

// ---------------------------------------------------------------------------
// Kernel 1: gate = relu(x@W1 + b1) @ W2 + b2, plus per-segment atomic max.
// 256 threads = 8 waves. Wave w owns HID column-tile w (16 cols). Grid-stride
// over 16-row M tiles. W1 fragments live in registers (swizzled once/block).
// ---------------------------------------------------------------------------
__global__ void __launch_bounds__(256)
gate_kernel(const float* __restrict__ x, const long long* __restrict__ index,
            const float* __restrict__ W1, const float* __restrict__ b1,
            const float* __restrict__ W2, const float* __restrict__ b2,
            float* __restrict__ gate_out, unsigned* __restrict__ smaxu) {
    __shared__ _Float16 wlds[32 * HID];     // 8 KB  (weight staging)
    __shared__ _Float16 xs[16 * IN_CH];     // 8 KB  (x tile, f16)
    __shared__ float    hbuf[16 * HID];     // 8 KB  (relu(h), row-major [row][col])
    __shared__ float    pbuf[16 * 16];      // 1 KB  (gate dot partials)
    __shared__ float    w2s[HID];
    __shared__ float    b2s;

    const int tid       = threadIdx.x;
    const int wave      = tid >> 5;
    const int lane      = tid & 31;
    const int colInTile = lane & 15;
    const int hiHalf    = lane >> 4;        // 0: lanes 0-15, 1: lanes 16-31

    if (tid < HID) w2s[tid] = W2[tid];
    if (tid == 0)  b2s = b2[0];

    // --- build register-resident B fragments of W1 (f16) ---
    const int nt   = wave;                  // ntile 0..7 (HID = 8 * 16)
    const int bcol = nt * 16 + colInTile;
    v16h breg[8];
    #pragma unroll
    for (int kt = 0; kt < 8; ++kt) {
        #pragma unroll
        for (int j = 0; j < 16; ++j) {      // stage 32xHID rows, coalesced
            int linear = j * 256 + tid;     // 0..4095
            wlds[linear] = (_Float16)W1[kt * 4096 + linear];
        }
        __syncthreads();
        const int r0 = hiHalf * 16;         // B: lanes<16 hold K=0..15, else 16..31
        #pragma unroll
        for (int j = 0; j < 16; ++j)
            breg[kt][j] = wlds[(r0 + j) * HID + bcol];
        __syncthreads();
    }
    const float b1v = b1[bcol];

    for (int tile = blockIdx.x; tile < M_TILES; tile += gridDim.x) {
        const int m0 = tile * 16;

        // --- prefetch next grid-stride tile's x slab (64 B per thread) ---
        const int nextTile = tile + gridDim.x;
        if (nextTile < M_TILES)
            __builtin_prefetch(x + (size_t)nextTile * 16 * IN_CH + tid * 16, 0, 1);

        // --- cooperative load of x tile, f32 -> f16 LDS ---
        const float4* x4 = reinterpret_cast<const float4*>(x + (size_t)m0 * IN_CH);
        #pragma unroll
        for (int j = 0; j < 4; ++j) {
            int q = j * 256 + tid;          // 1024 float4s
            float4 val = x4[q];
            int base = q * 4;
            xs[base + 0] = (_Float16)val.x; xs[base + 1] = (_Float16)val.y;
            xs[base + 2] = (_Float16)val.z; xs[base + 3] = (_Float16)val.w;
        }
        __syncthreads();

        // --- preload all 8 A fragments (pipelined ds_load_b128 pairs) ---
        const int arow = colInTile;         // A: M = lane&15
        v16h afrag[8];
        #pragma unroll
        for (int kt = 0; kt < 8; ++kt) {
            const halfx8* p =
                reinterpret_cast<const halfx8*>(&xs[arow * IN_CH + kt * 32 + hiHalf * 16]);
            halfx8 lo = p[0];               // K = {0..7} or {16..23}
            halfx8 hi = p[1];               // K = {8..15} or {24..31}
            #pragma unroll
            for (int j = 0; j < 8; ++j) { afrag[kt][j] = lo[j]; afrag[kt][8 + j] = hi[j]; }
        }

        // --- WMMA chain: h[16x16 tile] = x[16x256] @ W1[:, nt*16..+16] ---
        v8f acc = {};
        #pragma unroll
        for (int kt = 0; kt < 8; ++kt)
            acc = __builtin_amdgcn_wmma_f32_16x16x32_f16(
                false, afrag[kt], false, breg[kt], (short)0, acc, false, false);

        // --- relu(+b1) -> hbuf (row-major), C layout: M = v + 8*hiHalf ---
        #pragma unroll
        for (int v = 0; v < 8; ++v) {
            int M = v + hiHalf * 8;
            float hv = acc[v] + b1v;
            hbuf[M * HID + bcol] = hv > 0.0f ? hv : 0.0f;
        }
        __syncthreads();

        // --- gate = h @ W2 + b2: all 256 threads (16 rows x 16 k-stripes) ---
        {
            int r = tid >> 4;
            int chunk = tid & 15;
            float p = 0.0f;
            #pragma unroll
            for (int j = 0; j < 8; ++j) {
                int k = chunk + 16 * j;     // conflict-free stride across lanes
                p += hbuf[r * HID + k] * w2s[k];
            }
            pbuf[tid] = p;                  // [r][chunk]
        }
        __syncthreads();
        if (tid < 16) {
            float g = b2s;
            #pragma unroll
            for (int c2 = 0; c2 < 16; ++c2) g += pbuf[tid * 16 + c2];
            gate_out[m0 + tid] = g;
            long long s = index[m0 + tid];
            atomicMax(&smaxu[(int)s], flipf(g));
        }
        __syncthreads();
    }
}

// ---------------------------------------------------------------------------
// Kernel 2: decode segment max; empty / non-finite segments -> 0 (ref semantics)
// ---------------------------------------------------------------------------
__global__ void finalize_smax_kernel(const unsigned* __restrict__ smaxu,
                                     float* __restrict__ smaxf) {
    int i = blockIdx.x * blockDim.x + threadIdx.x;
    if (i >= SEGS) return;
    unsigned u = smaxu[i];
    float m = (u == 0u) ? 0.0f : unflipf(u);
    if (!isfinite(m)) m = 0.0f;
    smaxf[i] = m;
}

// ---------------------------------------------------------------------------
// Kernel 3: e = exp(gate - segmax[idx]); denom[idx] += e
// ---------------------------------------------------------------------------
__global__ void softmax_num_kernel(const float* __restrict__ gate,
                                   const long long* __restrict__ index,
                                   const float* __restrict__ smaxf,
                                   float* __restrict__ e_out,
                                   float* __restrict__ denom) {
    int i = blockIdx.x * blockDim.x + threadIdx.x;
    if (i >= N_ROWS) return;
    long long s = index[i];
    float e = expf(gate[i] - smaxf[(int)s]);
    e_out[i] = e;
    atomicAdd(&denom[(int)s], e);
}

// ---------------------------------------------------------------------------
// Kernel 4: t = relu(x@Wt + bt); out[idx] += alpha * t  (run-merged scatter)
// 256 threads = 8 waves; wave w owns OUT column-tiles {2w, 2w+1}.
// ---------------------------------------------------------------------------
__global__ void __launch_bounds__(256)
transform_kernel(const float* __restrict__ x, const long long* __restrict__ index,
                 const float* __restrict__ Wt, const float* __restrict__ bt,
                 const float* __restrict__ g_e, const float* __restrict__ g_denom,
                 float* __restrict__ out) {
    __shared__ _Float16 wlds[32 * OUT_CH];  // 16 KB (weight staging)
    __shared__ _Float16 xs[16 * IN_CH];     // 8 KB
    __shared__ float    tbuf[16 * OUT_CH];  // 16 KB (weighted t tile)
    __shared__ float    alpha_lds[16];
    __shared__ int      idx_lds[16];

    const int tid       = threadIdx.x;
    const int wave      = tid >> 5;
    const int lane      = tid & 31;
    const int colInTile = lane & 15;
    const int hiHalf    = lane >> 4;

    // --- register-resident B fragments of Wt: 2 ntiles x 8 k-steps ---
    v16h  breg[2][8];
    float btv[2];
    int   bcolq[2];
    #pragma unroll
    for (int q = 0; q < 2; ++q) {
        bcolq[q] = (2 * wave + q) * 16 + colInTile;
        btv[q]   = bt[bcolq[q]];
    }
    #pragma unroll
    for (int kt = 0; kt < 8; ++kt) {
        #pragma unroll
        for (int j = 0; j < 32; ++j) {      // stage 32x256 rows, coalesced
            int linear = j * 256 + tid;     // 0..8191
            wlds[linear] = (_Float16)Wt[kt * 8192 + linear];
        }
        __syncthreads();
        const int r0 = hiHalf * 16;
        #pragma unroll
        for (int q = 0; q < 2; ++q)
            #pragma unroll
            for (int j = 0; j < 16; ++j)
                breg[q][kt][j] = wlds[(r0 + j) * OUT_CH + bcolq[q]];
        __syncthreads();
    }

    for (int tile = blockIdx.x; tile < M_TILES; tile += gridDim.x) {
        const int m0 = tile * 16;

        // --- prefetch next grid-stride tile's x slab ---
        const int nextTile = tile + gridDim.x;
        if (nextTile < M_TILES)
            __builtin_prefetch(x + (size_t)nextTile * 16 * IN_CH + tid * 16, 0, 1);

        // --- load x tile (f32->f16) + per-row alpha / segment ids ---
        const float4* x4 = reinterpret_cast<const float4*>(x + (size_t)m0 * IN_CH);
        #pragma unroll
        for (int j = 0; j < 4; ++j) {
            int q = j * 256 + tid;
            float4 val = x4[q];
            int base = q * 4;
            xs[base + 0] = (_Float16)val.x; xs[base + 1] = (_Float16)val.y;
            xs[base + 2] = (_Float16)val.z; xs[base + 3] = (_Float16)val.w;
        }
        if (tid < 16) {
            long long s = index[m0 + tid];
            alpha_lds[tid] = g_e[m0 + tid] / fmaxf(g_denom[(int)s], 1e-16f);
            idx_lds[tid] = (int)s;
        }
        __syncthreads();

        // --- WMMA: 2 column tiles, K = 256; A preloaded in batches of 4 ---
        v8f acc0 = {}, acc1 = {};
        const int arow = colInTile;
        #pragma unroll
        for (int halfk = 0; halfk < 2; ++halfk) {
            v16h afrag[4];
            #pragma unroll
            for (int kk = 0; kk < 4; ++kk) {
                int kt = halfk * 4 + kk;
                const halfx8* p =
                    reinterpret_cast<const halfx8*>(&xs[arow * IN_CH + kt * 32 + hiHalf * 16]);
                halfx8 lo = p[0];
                halfx8 hi = p[1];
                #pragma unroll
                for (int j = 0; j < 8; ++j) { afrag[kk][j] = lo[j]; afrag[kk][8 + j] = hi[j]; }
            }
            #pragma unroll
            for (int kk = 0; kk < 4; ++kk) {
                int kt = halfk * 4 + kk;
                acc0 = __builtin_amdgcn_wmma_f32_16x16x32_f16(
                    false, afrag[kk], false, breg[0][kt], (short)0, acc0, false, false);
                acc1 = __builtin_amdgcn_wmma_f32_16x16x32_f16(
                    false, afrag[kk], false, breg[1][kt], (short)0, acc1, false, false);
            }
        }

        // --- relu(+bt) * alpha -> tbuf ---
        #pragma unroll
        for (int v = 0; v < 8; ++v) {
            int M = v + hiHalf * 8;
            float a0 = acc0[v] + btv[0];
            float a1 = acc1[v] + btv[1];
            a0 = a0 > 0.0f ? a0 : 0.0f;
            a1 = a1 > 0.0f ? a1 : 0.0f;
            float al = alpha_lds[M];
            tbuf[M * OUT_CH + bcolq[0]] = a0 * al;
            tbuf[M * OUT_CH + bcolq[1]] = a1 * al;
        }
        __syncthreads();

        // --- scatter: one column per thread, merge runs of equal index ---
        {
            const int col = tid;            // 256 cols, 256 threads
            float accum = 0.0f;
            #pragma unroll
            for (int r = 0; r < 16; ++r) {
                accum += tbuf[r * OUT_CH + col];
                if (r == 15 || idx_lds[r + 1] != idx_lds[r]) {
                    atomicAdd(&out[(size_t)idx_lds[r] * OUT_CH + col], accum);
                    accum = 0.0f;
                }
            }
        }
        __syncthreads();                    // protect idx_lds/tbuf for next tile
    }
}

// ---------------------------------------------------------------------------
// Host launcher. Input order: x, index, dim_size, W1, b1, W2, b2, Wt, bt.
// ---------------------------------------------------------------------------
extern "C" void kernel_launch(void* const* d_in, const int* in_sizes, int n_in,
                              void* d_out, int out_size, void* d_ws, size_t ws_size,
                              hipStream_t stream) {
    const float*      x     = (const float*)d_in[0];
    const long long*  index = (const long long*)d_in[1];   // int64 per reference
    const float*      W1    = (const float*)d_in[3];
    const float*      b1    = (const float*)d_in[4];
    const float*      W2    = (const float*)d_in[5];
    const float*      b2    = (const float*)d_in[6];
    const float*      Wt    = (const float*)d_in[7];
    const float*      bt    = (const float*)d_in[8];
    float*            out   = (float*)d_out;

    // Workspace layout (floats): gate[N] | e[N] | smaxu[S] | smaxf[S] | denom[S]
    float*    g_gate  = (float*)d_ws;
    float*    g_e     = g_gate + N_ROWS;
    unsigned* g_smaxu = (unsigned*)(g_e + N_ROWS);
    float*    g_smaxf = (float*)(g_smaxu + SEGS);
    float*    g_denom = g_smaxf + SEGS;

    init_kernel<<<4096, 256, 0, stream>>>(out, g_denom, g_smaxu);

    gate_kernel<<<2048, 256, 0, stream>>>(x, index, W1, b1, W2, b2,
                                          g_gate, g_smaxu);

    finalize_smax_kernel<<<(SEGS + 255) / 256, 256, 0, stream>>>(g_smaxu, g_smaxf);

    softmax_num_kernel<<<(N_ROWS + 255) / 256, 256, 0, stream>>>(
        g_gate, index, g_smaxf, g_e, g_denom);

    transform_kernel<<<2048, 256, 0, stream>>>(x, index, Wt, bt,
                                               g_e, g_denom, out);
}
